// CrossAttention_48473000903123
// MI455X (gfx1250) — matrix operand order repaired and checked by reading, not compiled, not run on hardware
//
#include <hip/hip_runtime.h>
#include <cstdint>

// ---------------- problem constants ----------------
#define BB 4
#define SS 2048
#define HH 16
#define DE 1024
#define DC 768
#define DH 64
#define MTOK (BB * SS) // 8192

typedef __attribute__((ext_vector_type(16))) _Float16 v16h;
typedef __attribute__((ext_vector_type(8)))  _Float16 v8h;
typedef __attribute__((ext_vector_type(8)))  float    v8f;

__device__ __forceinline__ v8f wmma_f16(v16h a, v16h b, v8f c) {
  // D = A(16x32 f16) * B(32x16 f16) + C(16x16 f32)
  return __builtin_amdgcn_wmma_f32_16x16x32_f16(
      false, a, false, b, (short)0, c, false, false);
}

// A-fragment: p = rowbase + k0 + (lane>>4)*8 ; covers k0+h8+{0..7}, k0+16+h8+{0..7}
__device__ __forceinline__ v16h make_frag_a(const _Float16* p) {
  v8h lo = *(const v8h*)(p);
  v8h hi = *(const v8h*)(p + 16);
  v16h r;
#pragma unroll
  for (int i = 0; i < 8; ++i) { r[i] = lo[i]; r[8 + i] = hi[i]; }
  return r;
}

// B-fragment: p = colbase + k0 + (lane>>4)*16 ; covers k0+h8+{0..7}, k0+16+h8+{0..7}
__device__ __forceinline__ v16h make_frag_b(const _Float16* p) {
  const int hb = ((threadIdx.x & 31) >> 4) << 3;
  v8h lo = *(const v8h*)(p - hb);
  v8h hi = *(const v8h*)(p + 16 - hb);
  v16h r;
#pragma unroll
  for (int i = 0; i < 8; ++i) { r[i] = lo[i]; r[8 + i] = hi[i]; }
  return r;
}

// ---- CDNA5 async global->LDS copy (ASYNCcnt-tracked), 16 bytes per lane ----
__device__ __forceinline__ void async_copy_b128(void* lds_ptr, const void* gptr) {
  unsigned lds_addr = (unsigned)(uintptr_t)lds_ptr;
  unsigned long long ga = (unsigned long long)(uintptr_t)gptr;
  asm volatile("global_load_async_to_lds_b128 %0, %1, off"
               :: "v"(lds_addr), "v"(ga) : "memory");
}
__device__ __forceinline__ void wait_async_le8() {
  asm volatile("s_wait_asynccnt 0x8" ::: "memory");
}
__device__ __forceinline__ void wait_async_0() {
  asm volatile("s_wait_asynccnt 0x0" ::: "memory");
}

// ---------------- elementwise cast ----------------
__global__ void cast_f32_f16_kernel(const float* __restrict__ in,
                                    _Float16* __restrict__ out, int n) {
  int i = blockIdx.x * 256 + threadIdx.x;
  if (i < n) out[i] = (_Float16)in[i];
}

// W [K][N] f32 row-major -> Wt [N][K] f16 row-major
__global__ void transpose_cast_kernel(const float* __restrict__ W,
                                      _Float16* __restrict__ Wt,
                                      int Krows, int Ncols) {
  int n = blockIdx.x * 16 + threadIdx.x;
  int k = blockIdx.y * 16 + threadIdx.y;
  if (n < Ncols && k < Krows)
    Wt[(long long)n * Krows + k] = (_Float16)W[(long long)k * Ncols + n];
}

// ---------------- generic WMMA GEMM (32x64 C-tile / wave, reg double-buffer) --
// A  : f16 [M][Kd] row-major ; Bt : f16 [N][Kd] row-major
// modes: 0 = f16 head-split [bh][s][64] (Q, scale folded)
//        1 = f16 head-split (K)
//        2 = f16 V-transposed [bh][64][S]
//        3 = f32 row-major [M][N]
__global__ void __launch_bounds__(128)
gemm_wmma_kernel(const _Float16* __restrict__ A,
                 const _Float16* __restrict__ Bt,
                 const float* __restrict__ bias,
                 _Float16* __restrict__ outh,
                 float* __restrict__ outf,
                 int N, int Kd, float scale, int mode) {
  const int lane   = threadIdx.x & 31;
  const int wave   = threadIdx.x >> 5;
  const int l16    = lane & 15;
  const int half8  = (lane >> 4) << 3;
  const int half16 = (lane >> 4) << 4;
  const int r0 = blockIdx.x * 128 + wave * 32;
  const int c0 = blockIdx.y * 64;

  v8f acc[2][4];
#pragma unroll
  for (int rb = 0; rb < 2; ++rb)
#pragma unroll
    for (int nc = 0; nc < 4; ++nc) acc[rb][nc] = (v8f){};

  const _Float16* ar0 = A + (long long)(r0 + l16) * Kd;
  const _Float16* ar1 = A + (long long)(r0 + 16 + l16) * Kd;
  const _Float16* br[4];
#pragma unroll
  for (int nc = 0; nc < 4; ++nc)
    br[nc] = Bt + (long long)(c0 + nc * 16 + l16) * Kd;

  // software pipeline: prefetch k, compute k while loading k+32
  v16h a0 = make_frag_a(ar0 + half8);
  v16h a1 = make_frag_a(ar1 + half8);
  v16h b0 = make_frag_b(br[0] + half16);
  v16h b1 = make_frag_b(br[1] + half16);
  v16h b2 = make_frag_b(br[2] + half16);
  v16h b3 = make_frag_b(br[3] + half16);

  for (int k = 0; k < Kd; k += 32) {
    v16h ca0 = a0, ca1 = a1, cb0 = b0, cb1 = b1, cb2 = b2, cb3 = b3;
    int kn = k + 32;
    if (kn < Kd) {
      a0 = make_frag_a(ar0 + kn + half8);
      a1 = make_frag_a(ar1 + kn + half8);
      b0 = make_frag_b(br[0] + kn + half16);
      b1 = make_frag_b(br[1] + kn + half16);
      b2 = make_frag_b(br[2] + kn + half16);
      b3 = make_frag_b(br[3] + kn + half16);
    }
    acc[0][0] = wmma_f16(ca0, cb0, acc[0][0]);
    acc[1][0] = wmma_f16(ca1, cb0, acc[1][0]);
    acc[0][1] = wmma_f16(ca0, cb1, acc[0][1]);
    acc[1][1] = wmma_f16(ca1, cb1, acc[1][1]);
    acc[0][2] = wmma_f16(ca0, cb2, acc[0][2]);
    acc[1][2] = wmma_f16(ca1, cb2, acc[1][2]);
    acc[0][3] = wmma_f16(ca0, cb3, acc[0][3]);
    acc[1][3] = wmma_f16(ca1, cb3, acc[1][3]);
  }

#pragma unroll
  for (int nc = 0; nc < 4; ++nc) {
    int col = c0 + nc * 16 + l16;
    float bv = bias[col];
#pragma unroll
    for (int rb = 0; rb < 2; ++rb) {
#pragma unroll
      for (int v = 0; v < 8; ++v) {
        int row = r0 + rb * 16 + v + half8;
        float val = (acc[rb][nc][v] + bv) * scale;
        if (mode == 3) {
          outf[(long long)row * N + col] = val;
        } else {
          int b = row >> 11, s = row & (SS - 1);
          int h = col >> 6, d = col & 63;
          int bh = b * HH + h;
          if (mode == 2)
            outh[(((long long)bh * DH + d) * SS) + s] = (_Float16)val;
          else
            outh[(((long long)bh * SS + s) * DH) + d] = (_Float16)val;
        }
      }
    }
  }
}

// ---------------- flash attention with async-LDS K/V staging ----------------
// Q : f16 [bh][S][64] (pre-scaled), K : f16 [bh][S][64], Vt : f16 [bh][64][S]
// O : f16 [b][S][1024]
__global__ void __launch_bounds__(128)
flash_attn_kernel(const _Float16* __restrict__ Q,
                  const _Float16* __restrict__ K,
                  const _Float16* __restrict__ Vt,
                  _Float16* __restrict__ O) {
  __shared__ __align__(16) _Float16 Ktile[2][64][64]; // [buf][key][d]
  __shared__ __align__(16) _Float16 Vtile[2][64][64]; // [buf][d][key]
  __shared__ __align__(16) _Float16 Pl[4][16][64];

  const int tid    = threadIdx.x;
  const int lane   = tid & 31;
  const int wave   = tid >> 5;
  const int l16    = lane & 15;
  const int half8  = (lane >> 4) << 3;
  const int half16 = (lane >> 4) << 4;
  const int bh = blockIdx.y;
  const int b  = bh >> 4;
  const int h  = bh & 15;
  const int q0 = blockIdx.x * 64 + wave * 16;

  const _Float16* Qb = Q  + ((long long)bh * SS + q0) * DH;
  const _Float16* Kb = K  + (long long)bh * SS * DH;
  const _Float16* Vb = Vt + (long long)bh * DH * SS;

  // stage one 64x64 K tile + 64x64 V tile (8 async b128 ops per thread)
  auto stage = [&](int kt, int buf) {
    const _Float16* kg = Kb + (long long)kt * 64 * DH;
    const _Float16* vg = Vb + kt * 64;
#pragma unroll
    for (int c = 0; c < 4; ++c) {
      int idx = c * 128 + tid;                 // 0..511
      int row = idx >> 3, col = (idx & 7) << 3;
      async_copy_b128(&Ktile[buf][row][col], kg + row * DH + col);
    }
#pragma unroll
    for (int c = 0; c < 4; ++c) {
      int idx = c * 128 + tid;
      int row = idx >> 3, col = (idx & 7) << 3;
      async_copy_b128(&Vtile[buf][row][col], vg + (long long)row * SS + col);
    }
  };

  // resident Q fragments (16 rows x 64 d)
  v16h aq0 = make_frag_a(Qb + (long long)l16 * DH + 0  + half8);
  v16h aq1 = make_frag_a(Qb + (long long)l16 * DH + 32 + half8);

  v8f oacc[4];
#pragma unroll
  for (int dc = 0; dc < 4; ++dc) oacc[dc] = (v8f){};
  float mrow[8], lrow[8];
#pragma unroll
  for (int v = 0; v < 8; ++v) { mrow[v] = -1e30f; lrow[v] = 0.0f; }

  const int NT = SS / 64;
  stage(0, 0);

  for (int kt = 0; kt < NT; ++kt) {
    const int cur = kt & 1;
    if (kt + 1 < NT) {           // overlap next tile's copy with this compute
      stage(kt + 1, cur ^ 1);
      wait_async_le8();          // oldest 8 = current tile (in-order retire)
    } else {
      wait_async_0();
    }
    __syncthreads();             // tile visible block-wide

    // ---- scores: S = Q * K^T (16 q-rows x 64 keys) ----
    v8f sacc[4];
#pragma unroll
    for (int c = 0; c < 4; ++c) {
      const _Float16* kp = &Ktile[cur][c * 16 + l16][0];
      v16h b0 = make_frag_b(kp + 0  + half16);
      v16h b1 = make_frag_b(kp + 32 + half16);
      v8f z = (v8f){};
      z = wmma_f16(aq0, b0, z);
      z = wmma_f16(aq1, b1, z);
      sacc[c] = z;
    }

    // ---- online softmax (row = v + 8*half, cols across 16 lanes) ----
#pragma unroll
    for (int v = 0; v < 8; ++v) {
      float mx = fmaxf(fmaxf(sacc[0][v], sacc[1][v]),
                       fmaxf(sacc[2][v], sacc[3][v]));
#pragma unroll
      for (int msk = 1; msk < 16; msk <<= 1)
        mx = fmaxf(mx, __shfl_xor(mx, msk, 32));
      float mnew  = fmaxf(mrow[v], mx);
      float alpha = __expf(mrow[v] - mnew);
      float rs = 0.0f;
#pragma unroll
      for (int c = 0; c < 4; ++c) {
        float p = __expf(sacc[c][v] - mnew);
        sacc[c][v] = p;
        rs += p;
      }
#pragma unroll
      for (int msk = 1; msk < 16; msk <<= 1)
        rs += __shfl_xor(rs, msk, 32);
      lrow[v] = lrow[v] * alpha + rs;
      mrow[v] = mnew;
#pragma unroll
      for (int dc = 0; dc < 4; ++dc) oacc[dc][v] *= alpha;
    }

    // ---- P (C-layout) -> LDS -> A-layout ----
#pragma unroll
    for (int c = 0; c < 4; ++c)
#pragma unroll
      for (int v = 0; v < 8; ++v)
        Pl[wave][v + half8][c * 16 + l16] = (_Float16)sacc[c][v];
    __syncthreads();

    v16h ap0 = make_frag_a(&Pl[wave][l16][0  + half8]);
    v16h ap1 = make_frag_a(&Pl[wave][l16][32 + half8]);

    // ---- O += P * V ----
#pragma unroll
    for (int dc = 0; dc < 4; ++dc) {
      const _Float16* vp = &Vtile[cur][dc * 16 + l16][0];
      v16h b0 = make_frag_b(vp + 0  + half16);
      v16h b1 = make_frag_b(vp + 32 + half16);
      oacc[dc] = wmma_f16(ap0, b0, oacc[dc]);
      oacc[dc] = wmma_f16(ap1, b1, oacc[dc]);
    }
    __syncthreads();  // LDS reads done before next iteration's writes
  }

  // ---- normalize + store (merge heads) ----
#pragma unroll
  for (int v = 0; v < 8; ++v) {
    float inv = 1.0f / lrow[v];
    int srow = q0 + v + half8;
#pragma unroll
    for (int dc = 0; dc < 4; ++dc) {
      int col = h * 64 + dc * 16 + l16;
      O[((long long)(b * SS + srow)) * DE + col] = (_Float16)(oacc[dc][v] * inv);
    }
  }
}

// ---------------- host launch ----------------
extern "C" void kernel_launch(void* const* d_in, const int* in_sizes, int n_in,
                              void* d_out, int out_size, void* d_ws, size_t ws_size,
                              hipStream_t stream) {
  (void)in_sizes; (void)n_in; (void)out_size; (void)ws_size;
  const float* x  = (const float*)d_in[0];
  const float* y  = (const float*)d_in[1];
  const float* Wq = (const float*)d_in[2];
  const float* bq = (const float*)d_in[3];
  const float* Wk = (const float*)d_in[4];
  const float* bk = (const float*)d_in[5];
  const float* Wv = (const float*)d_in[6];
  const float* bv = (const float*)d_in[7];
  const float* Wo = (const float*)d_in[8];
  const float* bo = (const float*)d_in[9];
  float* out = (float*)d_out;

  char* ws = (char*)d_ws;
  size_t off = 0;
  auto alloc = [&](size_t bytes) {
    char* p = ws + off;
    off = (off + bytes + 255) & ~(size_t)255;
    return p;
  };
  _Float16* x_h  = (_Float16*)alloc((size_t)MTOK * DE * 2);
  _Float16* y_h  = (_Float16*)alloc((size_t)MTOK * DC * 2);
  _Float16* Wqt  = (_Float16*)alloc((size_t)DE * DE * 2);
  _Float16* Wkt  = (_Float16*)alloc((size_t)DE * DC * 2);
  _Float16* Wvt  = (_Float16*)alloc((size_t)DE * DC * 2);
  _Float16* Wot  = (_Float16*)alloc((size_t)DE * DE * 2);
  _Float16* Qh   = (_Float16*)alloc((size_t)MTOK * DE * 2);
  _Float16* Kh   = (_Float16*)alloc((size_t)MTOK * DE * 2);
  _Float16* Vth  = (_Float16*)alloc((size_t)MTOK * DE * 2);
  _Float16* Oh   = (_Float16*)alloc((size_t)MTOK * DE * 2);

  // 1) casts
  {
    int nx = MTOK * DE, ny = MTOK * DC;
    cast_f32_f16_kernel<<<(nx + 255) / 256, 256, 0, stream>>>(x, x_h, nx);
    cast_f32_f16_kernel<<<(ny + 255) / 256, 256, 0, stream>>>(y, y_h, ny);
  }
  // 2) weight transposes
  {
    dim3 blk(16, 16);
    transpose_cast_kernel<<<dim3(DE / 16, DE / 16), blk, 0, stream>>>(Wq, Wqt, DE, DE);
    transpose_cast_kernel<<<dim3(DE / 16, DC / 16), blk, 0, stream>>>(Wk, Wkt, DC, DE);
    transpose_cast_kernel<<<dim3(DE / 16, DC / 16), blk, 0, stream>>>(Wv, Wvt, DC, DE);
    transpose_cast_kernel<<<dim3(DE / 16, DE / 16), blk, 0, stream>>>(Wo, Wot, DE, DE);
  }
  // 3) Q/K/V projections
  {
    dim3 grid(MTOK / 128, DE / 64);
    gemm_wmma_kernel<<<grid, 128, 0, stream>>>(x_h, Wqt, bq, Qh, nullptr,
                                               DE, DE, 0.125f, 0);
    gemm_wmma_kernel<<<grid, 128, 0, stream>>>(y_h, Wkt, bk, Kh, nullptr,
                                               DE, DC, 1.0f, 1);
    gemm_wmma_kernel<<<grid, 128, 0, stream>>>(y_h, Wvt, bv, Vth, nullptr,
                                               DE, DC, 1.0f, 2);
  }
  // 4) attention
  flash_attn_kernel<<<dim3(SS / 64, BB * HH), 128, 0, stream>>>(Qh, Kh, Vth, Oh);
  // 5) output projection (fp32 out)
  gemm_wmma_kernel<<<dim3(MTOK / 128, DE / 64), 128, 0, stream>>>(
      Oh, Wot, bo, nullptr, out, DE, DE, 1.0f, 3);
}
